// MultiLatentAttention_77833397338589
// MI455X (gfx1250) — matrix operand order repaired and checked by reading, not compiled, not run on hardware
//
#include <hip/hip_runtime.h>
#include <cstddef>

// ---------------------------------------------------------------------------
// MLA forward for MI455X (gfx1250, wave32, WMMA).
// bf16 operands / fp32 WMMA accumulation; async VMEM->LDS staging with
// double-buffered GEMM tiles (global_load_async_to_lds_b128 + ASYNCcnt).
// ---------------------------------------------------------------------------

typedef __bf16 bf16;
typedef __attribute__((ext_vector_type(16))) __bf16 v16bf;
typedef __attribute__((ext_vector_type(8)))  __bf16 bf16x8;
typedef __attribute__((ext_vector_type(8)))  float  v8f;

#define D_MODEL 2048
#define N_HEADS 16
#define D_HEAD 128
#define D_LATENT 512
#define D_ROPE 64
#define BATCH 2
#define SEQ 2048
#define MROWS (BATCH * SEQ)          // 4096
#define DQK (D_HEAD + D_ROPE)        // 192
#define KV_ROW (2 * D_HEAD + D_ROPE) // 320

// ------------------------------ helpers ------------------------------------

__device__ __forceinline__ bf16 f2bf(float f) {
  unsigned u = __builtin_bit_cast(unsigned, f);
  unsigned short h = (unsigned short)((u + 0x7FFFu + ((u >> 16) & 1u)) >> 16);
  return __builtin_bit_cast(bf16, h);
}
__device__ __forceinline__ float bf2f(bf16 h) {
  unsigned short u = __builtin_bit_cast(unsigned short, h);
  return __builtin_bit_cast(float, (unsigned)u << 16);
}
__device__ __forceinline__ void store_out(float* p, float v) { *p = v; }
__device__ __forceinline__ void store_out(bf16* p, float v)  { *p = f2bf(v); }

__device__ __forceinline__ v8f vzero8() {
  v8f z;
#pragma unroll
  for (int i = 0; i < 8; ++i) z[i] = 0.0f;
  return z;
}

// Async global->LDS 128-bit copy (CDNA5 GLOBAL_LOAD_ASYNC_TO_LDS_B128, GV
// mode). LDS byte address = low 32 bits of the generic pointer (ISA 10.2).
__device__ __forceinline__ void async_b128(const bf16* g, bf16* l) {
  unsigned la = (unsigned)(size_t)l;
  asm volatile("global_load_async_to_lds_b128 %0, %1, off"
               :: "v"(la), "v"(g) : "memory");
}
__device__ __forceinline__ void wait_async0() {
  asm volatile("s_wait_asynccnt 0x0" ::: "memory");
}

// Fragment from two contiguous 8-element (16B) chunks.
__device__ __forceinline__ v16bf load_frag2(const bf16* p0, const bf16* p1) {
  bf16x8 lo = *(const bf16x8*)p0;
  bf16x8 hi = *(const bf16x8*)p1;
  v16bf r;
#pragma unroll
  for (int i = 0; i < 8; ++i) { r[i] = lo[i]; r[i + 8] = hi[i]; }
  return r;
}
__device__ __forceinline__ v16bf load_frag16(const bf16* p) {
  return load_frag2(p, p + 8);
}

__device__ __forceinline__ v8f wmma_bf16(v16bf a, v16bf b, v8f c) {
  return __builtin_amdgcn_wmma_f32_16x16x32_bf16(false, a, false, b,
                                                 (short)0, c, false, false);
}

// --------------------- one-time conversion kernels -------------------------

__global__ __launch_bounds__(256)
void convert_bf16(const float* __restrict__ src, bf16* __restrict__ dst,
                  int n) {
  int i = blockIdx.x * 256 + threadIdx.x;
  if (i < n) dst[i] = f2bf(src[i]);
}

// WT[n][k] = W[k][n]  (reads coalesced over n; one-time scatter writes)
__global__ __launch_bounds__(256)
void convert_transpose(const float* __restrict__ W, bf16* __restrict__ WT,
                       int K, int N) {
  size_t gid = (size_t)blockIdx.x * 256 + threadIdx.x;
  if (gid >= (size_t)K * N) return;
  int n = (int)(gid % N), k = (int)(gid / N);
  WT[(size_t)n * K + k] = f2bf(W[gid]);
}

// ------------------------------ GEMM ---------------------------------------
// C[M,N] = A[M,K] * Bt[N,K]^T, all-bf16 operands.  M,N %128==0, K%32==0.
// 256 threads = 8 waves; block tile 128x128; wave tile 32x64 (2x4 WMMAs).
// Double-buffered LDS tiles filled by async VMEM->LDS copies.

template <typename CT>
__global__ __launch_bounds__(256)
void gemm_wmma(const bf16* __restrict__ A, const bf16* __restrict__ Bt,
               CT* __restrict__ C, int M, int N, int K) {
  __shared__ bf16 As[2][128 * 40];  // [m][k], stride 40 (80B, 16B-aligned)
  __shared__ bf16 Bs[2][128 * 40];  // [n][k]

  const int t    = threadIdx.x;
  const int m0   = blockIdx.y * 128;
  const int n0   = blockIdx.x * 128;
  const int lane = t & 31;
  const int w    = t >> 5;
  const int mbase = (w >> 1) * 32;
  const int nbase = (w & 1) * 64;
  const int half = lane >> 4;
  const int l16  = lane & 15;

  // tile staging: 128 rows x 32 bf16 = 4 b128 chunks/row; 2 chunks/thread
  const int srow = t >> 1;            // 0..127
  const int sch  = (t & 1) * 16;      // chunk pair base (elements)
  auto issue_tile = [&](int k0, int buf) {
#pragma unroll
    for (int c = 0; c < 2; ++c) {
      int ch = sch + c * 8;
      async_b128(A  + (size_t)(m0 + srow) * K + k0 + ch,
                 &As[buf][srow * 40 + ch]);
      async_b128(Bt + (size_t)(n0 + srow) * K + k0 + ch,
                 &Bs[buf][srow * 40 + ch]);
    }
  };

  v8f acc[2][4];
#pragma unroll
  for (int mi = 0; mi < 2; ++mi)
#pragma unroll
    for (int ni = 0; ni < 4; ++ni) acc[mi][ni] = vzero8();

  issue_tile(0, 0);
  wait_async0();
  __syncthreads();

  const int nk = K >> 5;
  for (int ki = 0; ki < nk; ++ki) {
    const int cur = ki & 1;
    if (ki + 1 < nk) issue_tile((ki + 1) << 5, cur ^ 1);

    v16bf afr[2];
#pragma unroll
    for (int mi = 0; mi < 2; ++mi) {
      const bf16* ap = &As[cur][(mbase + mi * 16 + l16) * 40];
      afr[mi] = load_frag16(ap + half * 16);
    }
#pragma unroll
    for (int ni = 0; ni < 4; ++ni) {
      v16bf bfr =
          load_frag16(&Bs[cur][(nbase + ni * 16 + l16) * 40 + half * 16]);
      acc[0][ni] = wmma_bf16(afr[0], bfr, acc[0][ni]);
      acc[1][ni] = wmma_bf16(afr[1], bfr, acc[1][ni]);
    }
    wait_async0();
    __syncthreads();
  }

#pragma unroll
  for (int mi = 0; mi < 2; ++mi)
#pragma unroll
    for (int ni = 0; ni < 4; ++ni)
#pragma unroll
      for (int r = 0; r < 8; ++r) {
        int row = m0 + mbase + mi * 16 + r + half * 8;
        int col = n0 + nbase + ni * 16 + l16;
        store_out(&C[(size_t)row * N + col], acc[mi][ni][r]);
      }
}

// ------------------------------ RoPE ---------------------------------------

__global__ __launch_bounds__(256)
void rope_kernel(bf16* __restrict__ qf, bf16* __restrict__ kf) {
  size_t gid = (size_t)blockIdx.x * blockDim.x + threadIdx.x;
  if (gid >= (size_t)MROWS * N_HEADS * 32) return;
  int    i = (int)(gid & 31);
  int    h = (int)((gid >> 5) & 15);
  size_t m = gid >> 9;
  int    s = (int)(m & (SEQ - 1));
  float freq = (float)s * __expf(-(float)i * 0.28782313662425574f);
  float sn, cs;
  __sincosf(freq, &sn, &cs);

  bf16* q = qf + (m * N_HEADS + h) * DQK + D_HEAD;
  float x1 = bf2f(q[i]), x2 = bf2f(q[i + 32]);
  q[i]      = f2bf(x1 * cs - x2 * sn);
  q[i + 32] = f2bf(x2 * cs + x1 * sn);

  bf16* k = kf + (m * N_HEADS + h) * KV_ROW + D_HEAD;
  x1 = bf2f(k[i]); x2 = bf2f(k[i + 32]);
  k[i]      = f2bf(x1 * cs - x2 * sn);
  k[i + 32] = f2bf(x2 * cs + x1 * sn);
}

// --------------------------- Flash attention -------------------------------
// grid (S/64, B*H); 128 threads = 4 waves; wave owns 16 query rows.
// Online-softmax streaming over 64-key tiles (causal). K tile arrives via
// async VMEM->LDS; V tile is transposed through VGPRs into [dv][key].

__global__ __launch_bounds__(128)
void mla_flash(const bf16* __restrict__ qf, const bf16* __restrict__ kvf,
               bf16* __restrict__ ctx) {
  __shared__ bf16 Kt[64 * 200];       // [key][d]   stride 200 (400B rows)
  __shared__ bf16 Vt[128 * 72];       // [dv][key]  stride 72  (144B rows)
  __shared__ bf16 Pt[4 * 16 * 72];    // per-wave [16 q][64 key], stride 72

  const int qb  = blockIdx.x;
  const int bh  = blockIdx.y;
  const int b   = bh >> 4, h = bh & 15;
  const int t   = threadIdx.x;
  const int lane = t & 31, w = t >> 5;
  const int half = lane >> 4, l16 = lane & 15;
  const int q0  = qb * 64;
  const size_t mq = (size_t)b * SEQ + q0;

  v16bf Qf[6];
  {
    const bf16* qrow = qf + ((mq + w * 16 + l16) * N_HEADS + h) * DQK;
#pragma unroll
    for (int kd = 0; kd < 6; ++kd)
      Qf[kd] = load_frag2(qrow + kd * 32 + half * 8,
                          qrow + kd * 32 + 16 + half * 8);
  }

  v8f O[8];
#pragma unroll
  for (int d = 0; d < 8; ++d) O[d] = vzero8();
  float mrow[8], lrow[8];
#pragma unroll
  for (int r = 0; r < 8; ++r) { mrow[r] = -1e30f; lrow[r] = 0.0f; }

  const float scale = 0.07216878364870323f;  // 1/sqrt(192)
  const bf16* kvb = kvf + ((size_t)b * SEQ * N_HEADS + h) * KV_ROW;

  for (int kb = 0; kb <= q0; kb += 64) {
    __syncthreads();
    // K tile 64x192 -> Kt: 64*24 b128 chunks / 128 thr = 12 async copies/thr
#pragma unroll
    for (int i = 0; i < 12; ++i) {
      int idx = i * 128 + t;
      int row = idx / 24, ch = idx % 24;
      async_b128(kvb + (size_t)(kb + row) * (N_HEADS * KV_ROW) + ch * 8,
                 &Kt[row * 200 + ch * 8]);
    }
    // V tile 64x128 -> Vt transposed [dv][key]
#pragma unroll
    for (int i = 0; i < 8; ++i) {
      int idx = i * 128 + t;
      int row = idx >> 4, ch = idx & 15;
      const bf16* src =
          kvb + (size_t)(kb + row) * (N_HEADS * KV_ROW) + DQK + ch * 8;
      bf16x8 vv = *(const bf16x8*)src;
#pragma unroll
      for (int j = 0; j < 8; ++j) Vt[(ch * 8 + j) * 72 + row] = vv[j];
    }
    wait_async0();
    __syncthreads();

    // scores: 16 q rows x 64 keys per wave (24 WMMAs)
    v8f sc[4];
#pragma unroll
    for (int nt = 0; nt < 4; ++nt) {
      sc[nt] = vzero8();
#pragma unroll
      for (int kd = 0; kd < 6; ++kd) {
        v16bf bfr =
            load_frag2(&Kt[(nt * 16 + l16) * 200 + kd * 32 + half * 8], &Kt[(nt * 16 + l16) * 200 + kd * 32 + 16 + half * 8]);
        sc[nt] = wmma_bf16(Qf[kd], bfr, sc[nt]);
      }
    }
    // scale + causal mask
#pragma unroll
    for (int nt = 0; nt < 4; ++nt) {
      int kn = kb + nt * 16 + l16;
#pragma unroll
      for (int r = 0; r < 8; ++r) {
        int qr = q0 + w * 16 + r + half * 8;
        float v = sc[nt][r] * scale;
        sc[nt][r] = (kn > qr) ? -1e30f : v;
      }
    }
    // online softmax (rows live across the 16 lanes of each half-wave)
#pragma unroll
    for (int r = 0; r < 8; ++r) {
      float mx = fmaxf(fmaxf(sc[0][r], sc[1][r]), fmaxf(sc[2][r], sc[3][r]));
#pragma unroll
      for (int off = 8; off >= 1; off >>= 1)
        mx = fmaxf(mx, __shfl_xor(mx, off, 32));
      float mnew = fmaxf(mrow[r], mx);
      float sf = __expf(mrow[r] - mnew);
      mrow[r] = mnew;
      float sum = 0.0f;
#pragma unroll
      for (int nt = 0; nt < 4; ++nt) {
        float p = __expf(sc[nt][r] - mnew);
        sum += p;
        Pt[w * 1152 + (r + half * 8) * 72 + nt * 16 + l16] = f2bf(p);
      }
#pragma unroll
      for (int off = 8; off >= 1; off >>= 1) sum += __shfl_xor(sum, off, 32);
      lrow[r] = lrow[r] * sf + sum;
#pragma unroll
      for (int d = 0; d < 8; ++d) O[d][r] *= sf;
    }
    // same-wave P write -> read ordering
    asm volatile("s_wait_dscnt 0x0" ::: "memory");

    // O += P(16x64) @ V(64x128)  (16 WMMAs)
#pragma unroll
    for (int kc = 0; kc < 64; kc += 32) {
      const bf16* pp = &Pt[w * 1152 + l16 * 72 + kc];
      v16bf pf = load_frag16(pp + half * 16);
#pragma unroll
      for (int d = 0; d < 8; ++d) {
        v16bf vf = load_frag16(&Vt[(d * 16 + l16) * 72 + kc + half * 16]);
        O[d] = wmma_bf16(pf, vf, O[d]);
      }
    }
  }

#pragma unroll
  for (int d = 0; d < 8; ++d)
#pragma unroll
    for (int r = 0; r < 8; ++r) {
      size_t row = mq + w * 16 + r + half * 8;
      ctx[row * D_MODEL + h * D_HEAD + d * 16 + l16] = f2bf(O[d][r] / lrow[r]);
    }
}

// ------------------------------ launcher -----------------------------------

extern "C" void kernel_launch(void* const* d_in, const int* in_sizes, int n_in,
                              void* d_out, int out_size, void* d_ws,
                              size_t ws_size, hipStream_t stream) {
  (void)in_sizes; (void)n_in; (void)out_size; (void)ws_size;
  const float* x    = (const float*)d_in[0];
  const float* Wkvd = (const float*)d_in[1];
  const float* Wkvu = (const float*)d_in[2];
  const float* Wqd  = (const float*)d_in[3];
  const float* Wqu  = (const float*)d_in[4];
  const float* Wo   = (const float*)d_in[5];
  float* out = (float*)d_out;

  char* ws = (char*)d_ws;
  const size_t MB = 1048576;
  bf16* kv_lat  = (bf16*)(ws);               //   4 MiB (4096 x 512)
  bf16* q_lat   = (bf16*)(ws + 4   * MB);    //   4 MiB
  bf16* kv_full = (bf16*)(ws + 8   * MB);    //  40 MiB (4096 x 5120)
  bf16* q_full  = (bf16*)(ws + 48  * MB);    //  24 MiB (4096 x 3072)
  bf16* ctx     = (bf16*)(ws + 72  * MB);    //  16 MiB (4096 x 2048)
  bf16* xb      = (bf16*)(ws + 88  * MB);    //  16 MiB (4096 x 2048)
  bf16* WkvdT   = (bf16*)(ws + 104 * MB);    //   2 MiB (512  x 2048)
  bf16* WkvuT   = (bf16*)(ws + 106 * MB);    //   5 MiB (5120 x 512)
  bf16* WqdT    = (bf16*)(ws + 111 * MB);    //   2 MiB (512  x 2048)
  bf16* WquT    = (bf16*)(ws + 113 * MB);    //   3 MiB (3072 x 512)
  bf16* WoT     = (bf16*)(ws + 116 * MB);    //   8 MiB (2048 x 2048)

  // one-time bf16 conversion / weight transposition
  {
    int n = MROWS * D_MODEL;
    convert_bf16<<<(n + 255) / 256, 256, 0, stream>>>(x, xb, n);
  }
  convert_transpose<<<(D_MODEL * D_LATENT + 255) / 256, 256, 0, stream>>>(
      Wkvd, WkvdT, D_MODEL, D_LATENT);
  convert_transpose<<<(D_LATENT * N_HEADS * KV_ROW + 255) / 256, 256, 0,
                      stream>>>(Wkvu, WkvuT, D_LATENT, N_HEADS * KV_ROW);
  convert_transpose<<<(D_MODEL * D_LATENT + 255) / 256, 256, 0, stream>>>(
      Wqd, WqdT, D_MODEL, D_LATENT);
  convert_transpose<<<(D_LATENT * N_HEADS * DQK + 255) / 256, 256, 0,
                      stream>>>(Wqu, WquT, D_LATENT, N_HEADS * DQK);
  convert_transpose<<<(D_MODEL * D_MODEL + 255) / 256, 256, 0, stream>>>(
      Wo, WoT, D_MODEL, D_MODEL);

  dim3 blk(256);
  gemm_wmma<bf16><<<dim3(D_LATENT / 128, MROWS / 128), blk, 0, stream>>>(
      xb, WkvdT, kv_lat, MROWS, D_LATENT, D_MODEL);
  gemm_wmma<bf16><<<dim3(D_LATENT / 128, MROWS / 128), blk, 0, stream>>>(
      xb, WqdT, q_lat, MROWS, D_LATENT, D_MODEL);
  gemm_wmma<bf16><<<dim3((N_HEADS * KV_ROW) / 128, MROWS / 128), blk, 0,
                    stream>>>(kv_lat, WkvuT, kv_full, MROWS, N_HEADS * KV_ROW,
                              D_LATENT);
  gemm_wmma<bf16><<<dim3((N_HEADS * DQK) / 128, MROWS / 128), blk, 0,
                    stream>>>(q_lat, WquT, q_full, MROWS, N_HEADS * DQK,
                              D_LATENT);
  rope_kernel<<<(MROWS * N_HEADS * 32) / 256, 256, 0, stream>>>(q_full,
                                                                kv_full);
  mla_flash<<<dim3(SEQ / 64, BATCH * N_HEADS), dim3(128), 0, stream>>>(
      q_full, kv_full, ctx);
  gemm_wmma<float><<<dim3(D_MODEL / 128, MROWS / 128), blk, 0, stream>>>(
      ctx, WoT, out, MROWS, D_MODEL, N_HEADS * D_HEAD);
}
